// DeepseekV32MLA_18545668784295
// MI455X (gfx1250) — matrix operand order repaired and checked by reading, not compiled, not run on hardware
//
#include <hip/hip_runtime.h>
#include <hip/hip_bf16.h>

// ---------------------------------------------------------------------------
// DeepSeek-V3.2 sparse MLA block for gfx1250 (MI455X), bf16 WMMA pipeline.
// ---------------------------------------------------------------------------

typedef __bf16 bf16_t;
typedef __attribute__((ext_vector_type(16))) __bf16 v16bf;
typedef __attribute__((ext_vector_type(8)))  float  v8f;

static constexpr int Bc   = 2;
static constexpr int Sc   = 2048;
static constexpr int HIDc = 2048;
static constexpr int Hc   = 16;
static constexpr int QLc  = 512;
static constexpr int KVLc = 256;
static constexpr int NOPEc = 128;
static constexpr int ROPEc = 64;
static constexpr int QKDc = 192;
static constexpr int VDc  = 128;
static constexpr int HIc  = 8;
static constexpr int DIc  = 128;
static constexpr int TOPKc = 512;
static constexpr float NEGV = -1e30f;
static constexpr int Mrows = Bc * Sc; // 4096

__device__ __forceinline__ bf16_t f2bf(float f) { return (bf16_t)f; }
__device__ __forceinline__ float4 ld4(const float* p) {
  return *reinterpret_cast<const float4*>(p);
}

// ---------------------------------------------------------------------------
// Generic GEMM: C[M,N] = A[M,K] * B[N,K]^T  (f32 in/out, bf16 WMMA compute)
// block = 256 threads = 8 waves, tile 128x128, k-step 32.
// Software-pipelined: next k-tile's global loads are issued into registers
// before the current tile's WMMAs, so LOADcnt drains under the matrix math.
// ---------------------------------------------------------------------------
__global__ __launch_bounds__(256)
void gemm_bf16_nt(const float* __restrict__ A, const float* __restrict__ Bw,
                  float* __restrict__ C, int M, int N, int K) {
  __shared__ bf16_t aT[128][32];
  __shared__ bf16_t bT[128][32];
  const int tid  = threadIdx.x;
  const int lane = tid & 31;
  const int wid  = tid >> 5;
  const int half16 = lane >> 4;   // 0: lanes 0-15, 1: lanes 16-31
  const int ln     = lane & 15;
  const int m0 = blockIdx.y * 128;
  const int n0 = blockIdx.x * 128;
  const int wm = (wid & 3) * 32;  // 4 waves along M
  const int wn = (wid >> 2) * 64; // 2 waves along N

  v8f acc[2][4];
#pragma unroll
  for (int i = 0; i < 2; ++i)
#pragma unroll
    for (int j = 0; j < 4; ++j) acc[i][j] = v8f{0,0,0,0,0,0,0,0};

  // per-thread staging registers: 4 float4 for A, 4 for B (one 128x32 tile each)
  float4 ra[4], rb[4];
  auto load_tile = [&](int kt) {
#pragma unroll
    for (int u = 0; u < 4; ++u) {
      const int t = tid + u * 256;
      const int r = t >> 3, c = (t & 7) * 4;
      ra[u] = ld4(&A[(size_t)(m0 + r) * K + kt + c]);
      if (n0 + r < N) rb[u] = ld4(&Bw[(size_t)(n0 + r) * K + kt + c]);
      else            rb[u] = float4{0.f, 0.f, 0.f, 0.f};
    }
  };
  auto store_tile = [&]() {
#pragma unroll
    for (int u = 0; u < 4; ++u) {
      const int t = tid + u * 256;
      const int r = t >> 3, c = (t & 7) * 4;
      aT[r][c + 0] = f2bf(ra[u].x); aT[r][c + 1] = f2bf(ra[u].y);
      aT[r][c + 2] = f2bf(ra[u].z); aT[r][c + 3] = f2bf(ra[u].w);
      bT[r][c + 0] = f2bf(rb[u].x); bT[r][c + 1] = f2bf(rb[u].y);
      bT[r][c + 2] = f2bf(rb[u].z); bT[r][c + 3] = f2bf(rb[u].w);
    }
  };

  load_tile(0);
  for (int kt = 0; kt < K; kt += 32) {
    store_tile();
    __syncthreads();
    if (kt + 32 < K) {
      load_tile(kt + 32);  // in flight while the WMMAs below execute
      if (kt + 64 < K)     // deep prefetch (global_prefetch_b8)
        __builtin_prefetch(&A[(size_t)(m0 + (tid >> 1)) * K + kt + 64], 0, 1);
    }

    // build fragments (A 16x32 / B 32x16, contiguous k = half16*16 + e in both)
    v16bf af[2], bfr[4];
#pragma unroll
    for (int i = 0; i < 2; ++i) {
      const int row = wm + i * 16 + ln;
      const int kb = half16 * 16;
#pragma unroll
      for (int e = 0; e < 8; ++e)  af[i][e] = aT[row][kb + e];
#pragma unroll
      for (int e = 8; e < 16; ++e) af[i][e] = aT[row][kb + e];
    }
#pragma unroll
    for (int j = 0; j < 4; ++j) {
      const int col = wn + j * 16 + ln;
      const int kb = half16 * 16;
#pragma unroll
      for (int e = 0; e < 16; ++e) bfr[j][e] = bT[col][kb + e];
    }
#pragma unroll
    for (int i = 0; i < 2; ++i)
#pragma unroll
      for (int j = 0; j < 4; ++j)
        acc[i][j] = __builtin_amdgcn_wmma_f32_16x16x32_bf16(
            false, af[i], false, bfr[j], (short)0, acc[i][j], false, false);
    __syncthreads();
  }

  // epilogue (C layout: M = e + 8*half16, N = ln)
#pragma unroll
  for (int i = 0; i < 2; ++i)
#pragma unroll
    for (int j = 0; j < 4; ++j) {
      const int gn = n0 + wn + j * 16 + ln;
      if (gn < N) {
#pragma unroll
        for (int e = 0; e < 8; ++e) {
          const int gm = m0 + wm + i * 16 + half16 * 8 + e;
          C[(size_t)gm * N + gn] = acc[i][j][e];
        }
      }
    }
}

// ---------------------------------------------------------------------------
// RMSNorm over rows (block per row)
// ---------------------------------------------------------------------------
__global__ __launch_bounds__(256)
void rmsnorm_kernel(const float* __restrict__ in, float* __restrict__ out,
                    const float* __restrict__ w, int L,
                    long long inPitch, long long outPitch, float eps) {
  __shared__ float red[256];
  const int row = blockIdx.x;
  const float* ip = in + (long long)row * inPitch;
  float* op = out + (long long)row * outPitch;
  float ss = 0.f;
  for (int i = threadIdx.x; i < L; i += 256) { float v = ip[i]; ss += v * v; }
  red[threadIdx.x] = ss; __syncthreads();
  for (int s = 128; s > 0; s >>= 1) {
    if (threadIdx.x < s) red[threadIdx.x] += red[threadIdx.x + s];
    __syncthreads();
  }
  const float inv = rsqrtf(red[0] / (float)L + eps);
  for (int i = threadIdx.x; i < L; i += 256) op[i] = ip[i] * inv * w[i];
}

// ---------------------------------------------------------------------------
// LayerNorm for DI=128 rows (wave per row)
// ---------------------------------------------------------------------------
__global__ __launch_bounds__(256)
void layernorm128_kernel(float* __restrict__ buf, const float* __restrict__ w,
                         const float* __restrict__ bb, int nRows, float eps) {
  const int row = blockIdx.x * 8 + (threadIdx.x >> 5);
  const int lane = threadIdx.x & 31;
  if (row >= nRows) return;
  float* p = buf + (long long)row * 128;
  float v[4]; float s = 0.f;
#pragma unroll
  for (int j = 0; j < 4; ++j) { v[j] = p[lane + 32 * j]; s += v[j]; }
  for (int m = 16; m > 0; m >>= 1) s += __shfl_xor(s, m, 32);
  const float mean = s / 128.f;
  float var = 0.f;
#pragma unroll
  for (int j = 0; j < 4; ++j) { float d = v[j] - mean; var += d * d; }
  for (int m = 16; m > 0; m >>= 1) var += __shfl_xor(var, m, 32);
  const float inv = rsqrtf(var / 128.f + eps);
#pragma unroll
  for (int j = 0; j < 4; ++j) {
    const int idx = lane + 32 * j;
    p[idx] = (v[j] - mean) * inv * w[idx] + bb[idx];
  }
}

// ---------------------------------------------------------------------------
// RoPE on a 64-wide slice embedded in rows (wave per row)
// ---------------------------------------------------------------------------
__global__ __launch_bounds__(256)
void rope_kernel(float* __restrict__ buf, const float* __restrict__ cosT,
                 const float* __restrict__ sinT, long long sliceOff,
                 long long pitch, int nRows, int posDiv) {
  const int row = blockIdx.x * 8 + (threadIdx.x >> 5);
  const int lane = threadIdx.x & 31;
  if (row >= nRows) return;
  const int pos = (row / posDiv) % Sc;
  float* p = buf + (long long)row * pitch + sliceOff;
  const float xr = p[2 * lane], xi = p[2 * lane + 1];
  const float c = cosT[pos * 32 + lane], s = sinT[pos * 32 + lane];
  p[2 * lane]     = xr * c - xi * s;
  p[2 * lane + 1] = xr * s + xi * c;
}

// ---------------------------------------------------------------------------
// Hadamard-128 rotate (wave per row): butterfly over 7 bits, scale 1/sqrt(128)
// ---------------------------------------------------------------------------
__global__ __launch_bounds__(256)
void hadamard128_kernel(float* __restrict__ buf, int nRows) {
  const int row = blockIdx.x * 8 + (threadIdx.x >> 5);
  const int lane = threadIdx.x & 31;
  if (row >= nRows) return;
  float* p = buf + (long long)row * 128;
  float v[4];
#pragma unroll
  for (int j = 0; j < 4; ++j) v[j] = p[lane + 32 * j];
  { // col bits 5,6 live in register index
    const float t0 = v[0] + v[1], t1 = v[0] - v[1];
    const float t2 = v[2] + v[3], t3 = v[2] - v[3];
    v[0] = t0 + t2; v[1] = t1 + t3; v[2] = t0 - t2; v[3] = t1 - t3;
  }
  for (int m = 1; m < 32; m <<= 1) {
#pragma unroll
    for (int j = 0; j < 4; ++j) {
      const float t = __shfl_xor(v[j], m, 32);
      v[j] = (lane & m) ? (t - v[j]) : (v[j] + t);
    }
  }
  const float sc = 0.08838834764831845f; // 128^-0.5
#pragma unroll
  for (int j = 0; j < 4; ++j) p[lane + 32 * j] = v[j] * sc;
}

// ---------------------------------------------------------------------------
// Indexer scores: scores[b,q,k] = sum_h relu(qi.ki)*w_idx + causal
// block = 8 waves, each wave owns one 16x16 (q,k) tile; shared ki tile.
// ---------------------------------------------------------------------------
__global__ __launch_bounds__(256)
void idx_scores_kernel(const float* __restrict__ qi, const float* __restrict__ ki,
                       const float* __restrict__ w_idx, float* __restrict__ scores) {
  __shared__ bf16_t kiT[16][128];
  const int b  = blockIdx.z;
  const int k0 = blockIdx.x * 16;
  const int tid = threadIdx.x, lane = tid & 31, wid = tid >> 5;
  const int half16 = lane >> 4, ln = lane & 15;

  for (int t = tid; t < 16 * 128; t += 256) {
    const int r = t >> 7, c = t & 127;
    kiT[r][c] = f2bf(ki[((size_t)(b * Sc + k0 + r)) * 128 + c]);
  }
  __syncthreads();

  const int q0 = (blockIdx.y * 8 + wid) * 16;
  if (k0 > q0 + 15) { // fully above diagonal
#pragma unroll
    for (int e = 0; e < 8; ++e)
      scores[((size_t)(b * Sc + q0 + half16 * 8 + e)) * Sc + k0 + ln] = NEGV;
    return;
  }

  v8f tot = v8f{0,0,0,0,0,0,0,0};
  for (int h = 0; h < HIc; ++h) {
    v8f c = v8f{0,0,0,0,0,0,0,0};
#pragma unroll
    for (int kc = 0; kc < 4; ++kc) {
      v16bf a, bv;
      const float* ap = &qi[(((size_t)(b * Sc + q0 + ln)) * HIc + h) * 128 + kc * 32];
      const int kb = half16 * 16;
      const float4 a0 = ld4(ap + kb),      a1 = ld4(ap + kb + 4);
      const float4 a2 = ld4(ap + kb + 8), a3 = ld4(ap + kb + 12);
      a[0]  = f2bf(a0.x); a[1]  = f2bf(a0.y); a[2]  = f2bf(a0.z); a[3]  = f2bf(a0.w);
      a[4]  = f2bf(a1.x); a[5]  = f2bf(a1.y); a[6]  = f2bf(a1.z); a[7]  = f2bf(a1.w);
      a[8]  = f2bf(a2.x); a[9]  = f2bf(a2.y); a[10] = f2bf(a2.z); a[11] = f2bf(a2.w);
      a[12] = f2bf(a3.x); a[13] = f2bf(a3.y); a[14] = f2bf(a3.z); a[15] = f2bf(a3.w);
      const int kb2 = half16 * 16;
#pragma unroll
      for (int e = 0; e < 16; ++e) bv[e] = kiT[ln][kc * 32 + kb2 + e];
      c = __builtin_amdgcn_wmma_f32_16x16x32_bf16(false, a, false, bv,
                                                  (short)0, c, false, false);
    }
#pragma unroll
    for (int e = 0; e < 8; ++e) {
      const float wv =
          w_idx[((size_t)(b * Sc + q0 + half16 * 8 + e)) * HIc + h] * 0.03125f;
      const float cv = c[e];
      tot[e] += (cv > 0.f ? cv : 0.f) * wv;
    }
  }
#pragma unroll
  for (int e = 0; e < 8; ++e) {
    const int q = q0 + half16 * 8 + e, k = k0 + ln;
    scores[((size_t)(b * Sc + q)) * Sc + k] = (k <= q) ? tot[e] : NEGV;
  }
}

// ---------------------------------------------------------------------------
// Top-k(512) per row via threshold binary search; rewrites row to {0, NEG}.
// ---------------------------------------------------------------------------
__global__ __launch_bounds__(256)
void topk_bias_kernel(float* __restrict__ scores) {
  __shared__ float vals[Sc];
  __shared__ float red[256];
  __shared__ int cnt;
  const int q = blockIdx.x, b = blockIdx.y, tid = threadIdx.x;
  float* row = scores + ((size_t)(b * Sc + q)) * Sc;
  float lmax = -3.4e38f, lmin = 3.4e38f;
  for (int k = tid; k < Sc; k += 256) {
    const float v = row[k]; vals[k] = v;
    lmax = fmaxf(lmax, v); lmin = fminf(lmin, v);
  }
  red[tid] = lmax; __syncthreads();
  for (int s = 128; s > 0; s >>= 1) {
    if (tid < s) red[tid] = fmaxf(red[tid], red[tid + s]); __syncthreads();
  }
  float hi = red[0]; __syncthreads();
  red[tid] = lmin; __syncthreads();
  for (int s = 128; s > 0; s >>= 1) {
    if (tid < s) red[tid] = fminf(red[tid], red[tid + s]); __syncthreads();
  }
  float lo = red[0]; __syncthreads();

  for (int it = 0; it < 30; ++it) {
    const float mid = 0.5f * (lo + hi);
    if (tid == 0) cnt = 0;
    __syncthreads();
    int lc = 0;
    for (int k = tid; k < Sc; k += 256) lc += (vals[k] >= mid) ? 1 : 0;
    atomicAdd(&cnt, lc);
    __syncthreads();
    const int c = cnt;
    if (c >= TOPKc) lo = mid; else hi = mid;
    __syncthreads();
  }
  const float t = lo;
  for (int k = tid; k < Sc; k += 256)
    row[k] = (vals[k] >= t && k <= q) ? 0.0f : NEGV;
}

// ---------------------------------------------------------------------------
// Causal flash attention with sparse bias.
// block = 8 waves for one (b,h,128-q-tile); each wave: 16 q rows x 32 k tiles.
// ---------------------------------------------------------------------------
__global__ __launch_bounds__(256)
void flash_attn_kernel(const float* __restrict__ qf, const float* __restrict__ kvb,
                       const float* __restrict__ kvr, const float* __restrict__ bias,
                       float* __restrict__ outp) {
  __shared__ bf16_t kf_lds[32][QKDc];   // [k][d]
  __shared__ bf16_t v_lds[VDc][32];     // [d][k] (transposed for PV B-fragments)
  __shared__ bf16_t p_lds[8][16 * 32];  // per-wave P staging
  const int b = blockIdx.z, h = blockIdx.y, qb0 = blockIdx.x * 128;
  const int tid = threadIdx.x, lane = tid & 31, wid = tid >> 5;
  const int half16 = lane >> 4, ln = lane & 15;
  const int qRow0 = qb0 + wid * 16;
  const float scale = 0.07216878364870323f; // 192^-0.5

  // preload q fragments for this wave's 16 rows (A layout, 6 k-chunks of 32)
  v16bf aq[6];
  {
    const int qrow = qRow0 + ln;
    const float* qp = &qf[(((size_t)(b * Sc + qrow)) * Hc + h) * QKDc];
    const int kb = half16 * 16;
#pragma unroll
    for (int kc = 0; kc < 6; ++kc) {
      const float4 q0 = ld4(qp + kc * 32 + kb);
      const float4 q1 = ld4(qp + kc * 32 + kb + 4);
      const float4 q2 = ld4(qp + kc * 32 + kb + 8);
      const float4 q3 = ld4(qp + kc * 32 + kb + 12);
      aq[kc][0]  = f2bf(q0.x); aq[kc][1]  = f2bf(q0.y);
      aq[kc][2]  = f2bf(q0.z); aq[kc][3]  = f2bf(q0.w);
      aq[kc][4]  = f2bf(q1.x); aq[kc][5]  = f2bf(q1.y);
      aq[kc][6]  = f2bf(q1.z); aq[kc][7]  = f2bf(q1.w);
      aq[kc][8]  = f2bf(q2.x); aq[kc][9]  = f2bf(q2.y);
      aq[kc][10] = f2bf(q2.z); aq[kc][11] = f2bf(q2.w);
      aq[kc][12] = f2bf(q3.x); aq[kc][13] = f2bf(q3.y);
      aq[kc][14] = f2bf(q3.z); aq[kc][15] = f2bf(q3.w);
    }
  }

  v8f o[8];
#pragma unroll
  for (int j = 0; j < 8; ++j) o[j] = v8f{0,0,0,0,0,0,0,0};
  float mrow[8], lrow[8];
#pragma unroll
  for (int e = 0; e < 8; ++e) { mrow[e] = -__builtin_inff(); lrow[e] = 0.f; }

  const int nkt = (qb0 + 128) >> 5;
  for (int kt = 0; kt < nkt; ++kt) {
    const int kBase = kt * 32;
    // ---- stage kf = [k_nope | k_pe] (32 x 192): batch loads, then convert
    float4 rkf[6];
#pragma unroll
    for (int u = 0; u < 6; ++u) {
      const int t = tid + u * 256;
      const int r = t / 48, c4 = t % 48;
      const int k = kBase + r;
      rkf[u] = (c4 < 32)
          ? ld4(&kvb[(((size_t)(b * Sc + k)) * Hc + h) * 256 + c4 * 4])
          : ld4(&kvr[((size_t)(b * Sc + k)) * 320 + 256 + (c4 - 32) * 4]);
    }
    // ---- stage V transposed (128 d x 32 k): batch loads, then convert
    float4 rv[4];
#pragma unroll
    for (int u = 0; u < 4; ++u) {
      const int t = tid + u * 256;
      const int kLoc = t & 31, d4 = t >> 5;
      rv[u] = ld4(&kvb[(((size_t)(b * Sc + kBase + kLoc)) * Hc + h) * 256 +
                       NOPEc + d4 * 4]);
    }
#pragma unroll
    for (int u = 0; u < 6; ++u) {
      const int t = tid + u * 256;
      const int r = t / 48, c = (t % 48) * 4;
      kf_lds[r][c + 0] = f2bf(rkf[u].x); kf_lds[r][c + 1] = f2bf(rkf[u].y);
      kf_lds[r][c + 2] = f2bf(rkf[u].z); kf_lds[r][c + 3] = f2bf(rkf[u].w);
    }
#pragma unroll
    for (int u = 0; u < 4; ++u) {
      const int t = tid + u * 256;
      const int kLoc = t & 31, d = (t >> 5) * 4;
      v_lds[d + 0][kLoc] = f2bf(rv[u].x);
      v_lds[d + 1][kLoc] = f2bf(rv[u].y);
      v_lds[d + 2][kLoc] = f2bf(rv[u].z);
      v_lds[d + 3][kLoc] = f2bf(rv[u].w);
    }
    __syncthreads();

    // S = qf . kf^T  (two 16x16 column tiles)
    v8f s0 = v8f{0,0,0,0,0,0,0,0}, s1 = v8f{0,0,0,0,0,0,0,0};
#pragma unroll
    for (int kc = 0; kc < 6; ++kc) {
      v16bf b0, b1;
      const int kb2 = half16 * 16;
#pragma unroll
      for (int e = 0; e < 16; ++e) {
        b0[e] = kf_lds[ln][kc * 32 + kb2 + e];
        b1[e] = kf_lds[16 + ln][kc * 32 + kb2 + e];
      }
      s0 = __builtin_amdgcn_wmma_f32_16x16x32_bf16(false, aq[kc], false, b0,
                                                   (short)0, s0, false, false);
      s1 = __builtin_amdgcn_wmma_f32_16x16x32_bf16(false, aq[kc], false, b1,
                                                   (short)0, s1, false, false);
    }

    // bias + online softmax update
    float p0[8], p1[8], sf[8];
#pragma unroll
    for (int e = 0; e < 8; ++e) {
      const int qrow = qRow0 + half16 * 8 + e;
      const size_t bb = ((size_t)(b * Sc + qrow)) * Sc + kBase;
      const float l0 = s0[e] * scale + bias[bb + ln];
      const float l1 = s1[e] * scale + bias[bb + 16 + ln];
      float mx = fmaxf(l0, l1);
      mx = fmaxf(mx, __shfl_xor(mx, 1, 32));
      mx = fmaxf(mx, __shfl_xor(mx, 2, 32));
      mx = fmaxf(mx, __shfl_xor(mx, 4, 32));
      mx = fmaxf(mx, __shfl_xor(mx, 8, 32));
      const float mn = fmaxf(mrow[e], mx);
      const float sfe = __expf(mrow[e] - mn); // exp(-inf)=0 on first tile
      const float e0 = __expf(l0 - mn), e1 = __expf(l1 - mn);
      float rs = e0 + e1;
      rs += __shfl_xor(rs, 1, 32);
      rs += __shfl_xor(rs, 2, 32);
      rs += __shfl_xor(rs, 4, 32);
      rs += __shfl_xor(rs, 8, 32);
      lrow[e] = lrow[e] * sfe + rs;
      mrow[e] = mn;
      p0[e] = e0; p1[e] = e1; sf[e] = sfe;
    }
#pragma unroll
    for (int j = 0; j < 8; ++j)
#pragma unroll
      for (int e = 0; e < 8; ++e) o[j][e] *= sf[e];

    // C-layout -> A-layout for P via per-wave LDS staging
    bf16_t* pp = p_lds[wid];
#pragma unroll
    for (int e = 0; e < 8; ++e) {
      const int r = half16 * 8 + e;
      pp[r * 32 + ln]      = f2bf(p0[e]);
      pp[r * 32 + 16 + ln] = f2bf(p1[e]);
    }
    asm volatile("s_wait_dscnt 0" ::: "memory"); // LDS in-order per wave; pin it
    v16bf pa;
    {
      const int r = ln, kb = half16 * 16;
#pragma unroll
      for (int e = 0; e < 8; ++e)  pa[e] = pp[r * 32 + kb + e];
#pragma unroll
      for (int e = 8; e < 16; ++e) pa[e] = pp[r * 32 + kb + e];
    }
    // O += P . V
#pragma unroll
    for (int j = 0; j < 8; ++j) {
      v16bf bv;
      const int n = j * 16 + ln, kb2 = half16 * 16;
#pragma unroll
      for (int e = 0; e < 16; ++e) bv[e] = v_lds[n][kb2 + e];
      o[j] = __builtin_amdgcn_wmma_f32_16x16x32_bf16(false, pa, false, bv,
                                                     (short)0, o[j], false, false);
    }
    __syncthreads();
  }

  // epilogue: out[b,q,h*128+d] = O / l
#pragma unroll
  for (int j = 0; j < 8; ++j)
#pragma unroll
    for (int e = 0; e < 8; ++e) {
      const int qrow = qRow0 + half16 * 8 + e;
      const int d = j * 16 + ln;
      outp[((size_t)(b * Sc + qrow)) * (Hc * VDc) + h * VDc + d] =
          o[j][e] / lrow[e];
    }
}

// ---------------------------------------------------------------------------
// Orchestration
// ---------------------------------------------------------------------------
extern "C" void kernel_launch(void* const* d_in, const int* in_sizes, int n_in,
                              void* d_out, int out_size, void* d_ws, size_t ws_size,
                              hipStream_t stream) {
  (void)in_sizes; (void)n_in; (void)out_size; (void)ws_size;
  const float* x          = (const float*)d_in[0];
  const float* cosT       = (const float*)d_in[1];
  const float* sinT       = (const float*)d_in[2];
  const float* wq_a       = (const float*)d_in[3];
  const float* q_a_ln_w   = (const float*)d_in[4];
  const float* wq_b       = (const float*)d_in[5];
  const float* wkv_a      = (const float*)d_in[6];
  const float* kv_a_ln_w  = (const float*)d_in[7];
  const float* wkv_b      = (const float*)d_in[8];
  const float* wo         = (const float*)d_in[9];
  const float* idx_wq_b   = (const float*)d_in[10];
  const float* idx_wk     = (const float*)d_in[11];
  const float* idx_k_nw   = (const float*)d_in[12];
  const float* idx_k_nb   = (const float*)d_in[13];
  const float* idx_w_proj = (const float*)d_in[14];
  float* out = (float*)d_out;

  float* ws = (float*)d_ws;
  size_t off = 0;
  auto alloc = [&](size_t n) { float* p = ws + off; off += n; return p; };
  float* q_resid = alloc((size_t)Mrows * QLc);        //  8 MB
  float* q       = alloc((size_t)Mrows * Hc * QKDc);  // 50 MB
  float* kv      = alloc((size_t)Mrows * 320);        //  5 MB
  float* ckvn    = alloc((size_t)Mrows * KVLc);       //  4 MB
  float* kvb     = alloc((size_t)Mrows * Hc * 256);   // 67 MB
  float* qi      = alloc((size_t)Mrows * HIc * DIc);  // 17 MB
  float* ki      = alloc((size_t)Mrows * DIc);        //  2 MB
  float* widx    = alloc((size_t)Mrows * HIc);        //  0.1 MB
  float* sc      = alloc((size_t)Bc * Sc * Sc);       // 34 MB
  float* ao      = alloc((size_t)Mrows * Hc * VDc);   // 34 MB

  const dim3 blk(256);
  auto gg = [](int N) { return dim3((unsigned)((N + 127) / 128), Mrows / 128); };

  // q path
  gemm_bf16_nt<<<gg(QLc), blk, 0, stream>>>(x, wq_a, q_resid, Mrows, QLc, HIDc);
  rmsnorm_kernel<<<Mrows, blk, 0, stream>>>(q_resid, q_resid, q_a_ln_w, QLc, QLc, QLc, 1e-6f);
  gemm_bf16_nt<<<gg(Hc * QKDc), blk, 0, stream>>>(q_resid, wq_b, q, Mrows, Hc * QKDc, QLc);
  rope_kernel<<<(Mrows * Hc) / 8, blk, 0, stream>>>(q, cosT, sinT, NOPEc, QKDc, Mrows * Hc, Hc);

  // kv path
  gemm_bf16_nt<<<gg(KVLc + ROPEc), blk, 0, stream>>>(x, wkv_a, kv, Mrows, KVLc + ROPEc, HIDc);
  rope_kernel<<<Mrows / 8, blk, 0, stream>>>(kv, cosT, sinT, KVLc, KVLc + ROPEc, Mrows, 1);
  rmsnorm_kernel<<<Mrows, blk, 0, stream>>>(kv, ckvn, kv_a_ln_w, KVLc, KVLc + ROPEc, KVLc, 1e-6f);
  gemm_bf16_nt<<<gg(Hc * 256), blk, 0, stream>>>(ckvn, wkv_b, kvb, Mrows, Hc * 256, KVLc);

  // indexer path
  gemm_bf16_nt<<<gg(HIc * DIc), blk, 0, stream>>>(q_resid, idx_wq_b, qi, Mrows, HIc * DIc, QLc);
  rope_kernel<<<(Mrows * HIc) / 8, blk, 0, stream>>>(qi, cosT, sinT, DIc - ROPEc, DIc, Mrows * HIc, HIc);
  gemm_bf16_nt<<<gg(DIc), blk, 0, stream>>>(x, idx_wk, ki, Mrows, DIc, HIDc);
  layernorm128_kernel<<<Mrows / 8, blk, 0, stream>>>(ki, idx_k_nw, idx_k_nb, Mrows, 1e-5f);
  rope_kernel<<<Mrows / 8, blk, 0, stream>>>(ki, cosT, sinT, DIc - ROPEc, DIc, Mrows, 1);
  hadamard128_kernel<<<(Mrows * HIc) / 8, blk, 0, stream>>>(qi, Mrows * HIc);
  hadamard128_kernel<<<Mrows / 8, blk, 0, stream>>>(ki, Mrows);
  gemm_bf16_nt<<<gg(HIc), blk, 0, stream>>>(x, idx_w_proj, widx, Mrows, HIc, HIDc);

  // scores -> top-k bias
  idx_scores_kernel<<<dim3(Sc / 16, Sc / (16 * 8), Bc), blk, 0, stream>>>(qi, ki, widx, sc);
  topk_bias_kernel<<<dim3(Sc, Bc), blk, 0, stream>>>(sc);

  // flash attention + output projection
  flash_attn_kernel<<<dim3(Sc / 128, Hc, Bc), blk, 0, stream>>>(q, kvb, kv, sc, ao);
  gemm_bf16_nt<<<gg(HIDc), blk, 0, stream>>>(ao, wo, out, Mrows, HIDc, Hc * VDc);
}
